// Map_9165460210124
// MI455X (gfx1250) — matrix operand order repaired and checked by reading, not compiled, not run on hardware
//
#include <hip/hip_runtime.h>
#include <math.h>

// ---------------- problem constants ----------------
#define L_LEV   16
#define F_FEAT  2
#define T_TAB   524288            // 2^19
#define M_SAMP  128
#define N_RAYS  8192
#define NPTS    (N_RAYS * M_SAMP) // 1,048,576

typedef __attribute__((ext_vector_type(16))) _Float16 v16h;
typedef __attribute__((ext_vector_type(8)))  float    v8f;

union V16HU { v16h v; uint4 q[2]; };

// floor(16 * (32)^(l/15)) = floor(16 * 2^(l/3))
__device__ __constant__ float kRes[16] = {
    16.f, 20.f, 25.f, 32.f, 40.f, 50.f, 64.f, 80.f,
    101.f, 128.f, 161.f, 203.f, 256.f, 322.f, 406.f, 512.f
};

// ---------------- wave-local LDS ordering ----------------
__device__ __forceinline__ void wave_sync() {
    __builtin_amdgcn_wave_barrier();
    asm volatile("s_wait_dscnt 0" ::: "memory");
    __builtin_amdgcn_wave_barrier();
}

// ---------------- WMMA helpers (CDNA5 16x16x32 f16 layouts) ----------------
// A matrix 16x32 f16: lane l holds row m = l&15.
//   lanes 0-15 : VGPR0..3 -> K=0..7,  VGPR4..7 -> K=16..23
//   lanes 16-31: VGPR0..3 -> K=8..15, VGPR4..7 -> K=24..31
__device__ __forceinline__ v16h lds_load_A(const _Float16* tile, int lane,
                                           int kTile, int strideHalfs) {
    const int m  = lane & 15;
    const int k0 = (lane & 16) ? 8 : 0;
    const _Float16* base = tile + m * strideHalfs + kTile * 32 + k0;
    V16HU u;
    u.q[0] = *(const uint4*)(base);        // K = k0 .. k0+7
    u.q[1] = *(const uint4*)(base + 16);   // K = 16+k0 .. 16+k0+7
    return u.v;
}

// B matrix 32x16 f16, stored in LDS as W[N][K] (i.e. out-feature major, so
// lane with column n reads W[n][k] which is exactly B[k][n]).
//   lanes 0-15 : n = lane,    K = 0..7,  16..23
//   lanes 16-31: n = lane-16, K = 8..15, 24..31
__device__ __forceinline__ v16h lds_load_B(const _Float16* w, int lane,
                                           int nTile, int kTile, int strideHalfs) {
    const int n  = nTile * 16 + (lane & 15);
    const int k0 = kTile * 32 + ((lane & 16) ? 8 : 0);
    const uint4* p = (const uint4*)(w + n * strideHalfs + k0);
    V16HU u;
    u.q[0] = p[0];
    u.q[1] = p[2];
    return u.v;
}

// C/D 16x16 f32: lane l holds column n = l&15, VGPR r holds row r + (l<16?0:8)
__device__ __forceinline__ v8f bias_row(const float* bias, int nTile, int lane) {
    const float bv = bias[nTile * 16 + (lane & 15)];
    v8f c;
#pragma unroll
    for (int r = 0; r < 8; ++r) c[r] = bv;
    return c;
}

__device__ __forceinline__ v8f wmma_f16(v16h a, v16h b, v8f c) {
    return __builtin_amdgcn_wmma_f32_16x16x32_f16(
        /*neg_a=*/false, a, /*neg_b=*/false, b,
        /*c_mod=*/(short)0, c, /*reuse_a=*/false, /*reuse_b=*/false);
}

__device__ __forceinline__ void store_relu_h16(v8f d, _Float16* buf, int lane,
                                               int nTile, int strideHalfs) {
    const int n  = nTile * 16 + (lane & 15);
    const int rb = (lane & 16) ? 8 : 0;
#pragma unroll
    for (int r = 0; r < 8; ++r) {
        float v = d[r];
        v = v > 0.0f ? v : 0.0f;
        buf[(rb + r) * strideHalfs + n] = (_Float16)v;
    }
}

__device__ __forceinline__ void store_sig_f32(v8f d, float* buf, int lane) {
    const int n  = lane & 15;
    const int rb = (lane & 16) ? 8 : 0;
#pragma unroll
    for (int r = 0; r < 8; ++r) {
        buf[(rb + r) * 16 + n] = 1.0f / (1.0f + __expf(-d[r]));
    }
}

// ---------------- kernel 0: per-ray directional encoding ----------------
__launch_bounds__(256)
__global__ void denc_kernel(const float* __restrict__ D, float* __restrict__ denc) {
    const int r = blockIdx.x * blockDim.x + threadIdx.x;
    if (r >= N_RAYS) return;
    const float d0 = D[3 * r + 0], d1 = D[3 * r + 1], d2 = D[3 * r + 2];
    float* o = denc + r * 24;
#pragma unroll
    for (int i = 0; i < 4; ++i) {
        const float s = (float)(1 << i) * 3.14159265358979323846f;
        o[i * 6 + 0] = __sinf(s * d0);
        o[i * 6 + 1] = __sinf(s * d1);
        o[i * 6 + 2] = __sinf(s * d2);
        o[i * 6 + 3] = __cosf(s * d0);
        o[i * 6 + 4] = __cosf(s * d1);
        o[i * 6 + 5] = __cosf(s * d2);
    }
}

// ---------------- kernel 1: fused hash-encode + MLPs ----------------
__launch_bounds__(128)
__global__ void nerf_fused(const float* __restrict__ X,  const float* __restrict__ tables,
                           const float* __restrict__ W1, const float* __restrict__ b1,
                           const float* __restrict__ W2, const float* __restrict__ b2,
                           const float* __restrict__ Wc1, const float* __restrict__ bc1,
                           const float* __restrict__ Wc2, const float* __restrict__ bc2,
                           const float* __restrict__ Wc3, const float* __restrict__ bc3,
                           const float* __restrict__ denc,
                           float* __restrict__ density, float* __restrict__ colour) {
    __shared__ __align__(16) _Float16 encS[128 * 32];   // per-point hash features
    __shared__ __align__(16) _Float16 W1h [64 * 32];    // [N][K]
    __shared__ __align__(16) _Float16 W2h [16 * 64];
    __shared__ __align__(16) _Float16 Wc1h[64 * 64];    // K padded 39 -> 64
    __shared__ __align__(16) _Float16 Wc2h[64 * 64];
    __shared__ __align__(16) _Float16 Wc3h[16 * 64];    // N padded 3 -> 16
    __shared__ float b1s[64], b2s[16], bc1s[64], bc2s[64], bc3s[16];
    __shared__ __align__(16) _Float16 bufA[4][16 * 64]; // per-wave activation ping
    __shared__ __align__(16) _Float16 bufB[4][16 * 64]; // per-wave activation pong
    __shared__ __align__(16) float    h2s [4][16 * 16]; // per-wave f32 staging

    const int tid  = threadIdx.x;
    const int lane = tid & 31;
    const int wave = tid >> 5;

    // ---- stage weights to LDS as f16 ----
    for (int i = tid; i < 64 * 32; i += 128) W1h[i] = (_Float16)W1[i];
    for (int i = tid; i < 16 * 64; i += 128) W2h[i] = (_Float16)W2[i];
    for (int i = tid; i < 64 * 64; i += 128) {
        const int n = i >> 6, k = i & 63;
        Wc1h[i] = (_Float16)((k < 39) ? Wc1[n * 39 + k] : 0.0f);
    }
    for (int i = tid; i < 64 * 64; i += 128) Wc2h[i] = (_Float16)Wc2[i];
    for (int i = tid; i < 16 * 64; i += 128) {
        const int n = i >> 6, k = i & 63;
        Wc3h[i] = (_Float16)((n < 3) ? Wc3[n * 64 + k] : 0.0f);
    }
    if (tid < 64) { b1s[tid] = b1[tid]; bc1s[tid] = bc1[tid]; bc2s[tid] = bc2[tid]; }
    if (tid < 16) { b2s[tid] = b2[tid]; bc3s[tid] = (tid < 3) ? bc3[tid] : 0.0f; }

    // ---- hash encode: one point per thread ----
    const int p = blockIdx.x * 128 + tid;
    {
        const float x = X[3 * p + 0], y = X[3 * p + 1], z = X[3 * p + 2];
        for (int l = 0; l < L_LEV; ++l) {
            const float r = kRes[l];
            const float fx = x * r, fy = y * r, fz = z * r;
            const float fx0 = floorf(fx), fy0 = floorf(fy), fz0 = floorf(fz);
            const float wx = fx - fx0, wy = fy - fy0, wz = fz - fz0;
            const unsigned cx = (unsigned)fx0, cy = (unsigned)fy0, cz = (unsigned)fz0;
            const float2* tab = ((const float2*)tables) + (size_t)l * T_TAB;
            float e0 = 0.0f, e1 = 0.0f;
#pragma unroll
            for (int c = 0; c < 8; ++c) {
                const unsigned ox = (c >> 2) & 1, oy = (c >> 1) & 1, oz = c & 1;
                const unsigned h = (cx + ox)
                                 ^ ((cy + oy) * 2654435761u)
                                 ^ ((cz + oz) * 805459861u);
                const unsigned idx = h & (T_TAB - 1);
                const float2 f = tab[idx];
                const float wt = (ox ? wx : 1.0f - wx)
                               * (oy ? wy : 1.0f - wy)
                               * (oz ? wz : 1.0f - wz);
                e0 += wt * f.x;
                e1 += wt * f.y;
            }
            encS[tid * 32 + 2 * l + 0] = (_Float16)e0;
            encS[tid * 32 + 2 * l + 1] = (_Float16)e1;
        }
    }
    __syncthreads();

    _Float16* bA = bufA[wave];
    _Float16* bB = bufB[wave];
    float*    h2 = h2s[wave];

    // each wave owns 32 consecutive points -> two 16-row WMMA tiles
    for (int t = 0; t < 2; ++t) {
        const int rowBase = wave * 32 + t * 16;
        const int pBase   = blockIdx.x * 128 + rowBase;
        const _Float16* encT = encS + rowBase * 32;

        // ---- layer 1: h1 = relu(enc @ W1^T + b1), 16x64, K=32 ----
        {
            const v16h a = lds_load_A(encT, lane, 0, 32);
#pragma unroll
            for (int n = 0; n < 4; ++n) {
                const v16h b = lds_load_B(W1h, lane, n, 0, 32);
                v8f c = bias_row(b1s, n, lane);
                c = wmma_f16(a, b, c);
                store_relu_h16(c, bA, lane, n, 64);
            }
        }
        wave_sync();

        // ---- layer 2: h2 = sigmoid(h1 @ W2^T + b2), 16x16, K=64 ----
        {
            const v16h a0 = lds_load_A(bA, lane, 0, 64);
            const v16h a1 = lds_load_A(bA, lane, 1, 64);
            v8f c = bias_row(b2s, 0, lane);
            c = wmma_f16(a0, lds_load_B(W2h, lane, 0, 0, 64), c);
            c = wmma_f16(a1, lds_load_B(W2h, lane, 0, 1, 64), c);
            store_sig_f32(c, h2, lane);
        }
        wave_sync();

        // ---- density out + build colour-MLP input row (39 used, pad 64) ----
        if (lane < 16) {
            const int m = lane;
            float hv = h2[m * 16];
            hv = fminf(fmaxf(hv, -15.0f), 15.0f);
            density[pBase + m] = __expf(hv);

            _Float16* row = bB + m * 64;
#pragma unroll
            for (int k = 0; k < 15; ++k) row[k] = (_Float16)h2[m * 16 + 1 + k];
            const float* de = denc + (size_t)((unsigned)(pBase + m) & (N_RAYS - 1)) * 24;
#pragma unroll
            for (int k = 0; k < 24; ++k) row[15 + k] = (_Float16)de[k];
#pragma unroll
            for (int k = 39; k < 64; ++k) row[k] = (_Float16)0.0f;
        }
        wave_sync();

        // ---- colour layer 1: relu(cin @ Wc1^T + bc1), 16x64, K=64 ----
        {
            const v16h a0 = lds_load_A(bB, lane, 0, 64);
            const v16h a1 = lds_load_A(bB, lane, 1, 64);
#pragma unroll
            for (int n = 0; n < 4; ++n) {
                v8f c = bias_row(bc1s, n, lane);
                c = wmma_f16(a0, lds_load_B(Wc1h, lane, n, 0, 64), c);
                c = wmma_f16(a1, lds_load_B(Wc1h, lane, n, 1, 64), c);
                store_relu_h16(c, bA, lane, n, 64);
            }
        }
        wave_sync();

        // ---- colour layer 2: relu(c1 @ Wc2^T + bc2), 16x64, K=64 ----
        {
            const v16h a0 = lds_load_A(bA, lane, 0, 64);
            const v16h a1 = lds_load_A(bA, lane, 1, 64);
#pragma unroll
            for (int n = 0; n < 4; ++n) {
                v8f c = bias_row(bc2s, n, lane);
                c = wmma_f16(a0, lds_load_B(Wc2h, lane, n, 0, 64), c);
                c = wmma_f16(a1, lds_load_B(Wc2h, lane, n, 1, 64), c);
                store_relu_h16(c, bB, lane, n, 64);
            }
        }
        wave_sync();

        // ---- colour layer 3: sigmoid(c2 @ Wc3^T + bc3), 16x16 (3 valid) ----
        {
            const v16h a0 = lds_load_A(bB, lane, 0, 64);
            const v16h a1 = lds_load_A(bB, lane, 1, 64);
            v8f c = bias_row(bc3s, 0, lane);
            c = wmma_f16(a0, lds_load_B(Wc3h, lane, 0, 0, 64), c);
            c = wmma_f16(a1, lds_load_B(Wc3h, lane, 0, 1, 64), c);
            store_sig_f32(c, h2, lane);
        }
        wave_sync();

        if (lane < 16) {
            const int m = lane;
            float* co = colour + (size_t)(pBase + m) * 3;
            co[0] = h2[m * 16 + 0];
            co[1] = h2[m * 16 + 1];
            co[2] = h2[m * 16 + 2];
        }
        wave_sync();
    }
}

// ---------------- launcher ----------------
extern "C" void kernel_launch(void* const* d_in, const int* in_sizes, int n_in,
                              void* d_out, int out_size, void* d_ws, size_t ws_size,
                              hipStream_t stream) {
    const float* X   = (const float*)d_in[0];
    const float* Dv  = (const float*)d_in[1];
    const float* tab = (const float*)d_in[2];
    const float* W1  = (const float*)d_in[3];
    const float* b1  = (const float*)d_in[4];
    const float* W2  = (const float*)d_in[5];
    const float* b2  = (const float*)d_in[6];
    const float* Wc1 = (const float*)d_in[7];
    const float* bc1 = (const float*)d_in[8];
    const float* Wc2 = (const float*)d_in[9];
    const float* bc2 = (const float*)d_in[10];
    const float* Wc3 = (const float*)d_in[11];
    const float* bc3 = (const float*)d_in[12];

    float* denc    = (float*)d_ws;           // 8192 * 24 floats = 768 KB
    float* density = (float*)d_out;          // [NPTS]
    float* colour  = density + NPTS;         // [NPTS * 3]

    denc_kernel<<<dim3(N_RAYS / 256), dim3(256), 0, stream>>>(Dv, denc);
    nerf_fused<<<dim3(NPTS / 128), dim3(128), 0, stream>>>(
        X, tab, W1, b1, W2, b2, Wc1, bc1, Wc2, bc2, Wc3, bc3,
        denc, density, colour);
}
